// CSNeuralODE_6459630813866
// MI455X (gfx1250) — matrix operand order repaired and checked by reading, not compiled, not run on hardware
//
#include <hip/hip_runtime.h>

#define NBATCH 65536
#define TSTEPS 500

typedef _Float16 v16h __attribute__((ext_vector_type(16)));
typedef float    v8f  __attribute__((ext_vector_type(8)));
typedef float    v2f  __attribute__((ext_vector_type(2)));

// ---- small helpers -------------------------------------------------------

static __device__ __forceinline__ unsigned pk2(float a, float b) {
  // v_cvt_pk_rtz_f16_f32: a -> [15:0], b -> [31:16]
  auto p = __builtin_amdgcn_cvt_pkrtz(a, b);
  unsigned u;
  __builtin_memcpy(&u, &p, 4);
  return u;
}

static __device__ __forceinline__ float act_tanh(float x) {
#if __has_builtin(__builtin_amdgcn_tanhf)
  return __builtin_amdgcn_tanhf(x);          // v_tanh_f32 (confirmed on gfx1250)
#elif __has_builtin(__builtin_amdgcn_tanh_f32)
  return __builtin_amdgcn_tanh_f32(x);
#else
  float e = __builtin_amdgcn_exp2f(x * 2.8853900817779268f);
  return 1.0f - 2.0f / (e + 1.0f);
#endif
}

static __device__ __forceinline__ float act_softplus(float x) {
  // ln(1+e^x) = ln2 * log2(1 + 2^(x*log2(e)))
  float e = __builtin_amdgcn_exp2f(x * 1.4426950408889634f);
  return 0.6931471805599453f * __builtin_amdgcn_logf(e + 1.0f);
}

static __device__ __forceinline__ v8f wmma_f16(v16h a, v16h b, v8f c) {
  return __builtin_amdgcn_wmma_f32_16x16x32_f16(false, a, false, b,
                                                (short)0, c, false, false);
}

static __device__ __forceinline__ v8f wmma_f32k4(v2f a, v2f b, v8f c) {
  // V_WMMA_F32_16X16X4_F32: A,B = 2 VGPRs each (f32), K = 4
  return __builtin_amdgcn_wmma_f32_16x16x4_f32(false, a, false, b,
                                               (short)0, c, false, false);
}

union BU { v16h h; unsigned u[8]; };

// Swizzle-free B packing. Two adjacent f32 D-tiles (hidden rows 32h..32h+31,
// N = batch) -> one f16 B operand for K = 32h..32h+31. With
//   b.u[j]   = pk(d0[2j], d0[2j+1])   (j = 0..3)
//   b.u[j+4] = pk(d1[2j], d1[2j+1])
// the hidden index delivered at logical K (AMD table order) is K itself;
// the A operands use the same order
// with unremapped columns (buildA_perm), so NO cross-lane ops
// are needed per step.
static __device__ __forceinline__ v16h packB(v8f d0, v8f d1) {
  BU b;
#pragma unroll
  for (int j = 0; j < 4; ++j) {
    b.u[j]     = pk2(d0[2 * j], d0[2 * j + 1]);
    b.u[j + 4] = pk2(d1[2 * j], d1[2 * j + 1]);
  }
  return b.h;
}

// ---- register-resident network tiles ------------------------------------

struct NetCore {
  v2f  A1[4];    // layer1: f32 16x16x4 A tiles: VGPR0 = w0|bias, VGPR1 = w1|0
  v16h A2[4][2]; // layer2: W2 [Mtile][Khalf], K columns in packB order (unpermuted)
};

// f16 A 16x32 layout: lane l holds row M = rowBase + (l%16);
// elements 0..7 -> K = kb..kb+7, elements 8..15 -> K = kb+16..kb+23,
// kb = 32*kHalf + (l>=16 ? 8 : 0). Column index is not remapped; this already
// matches packB's delivery order.
static __device__ __forceinline__ v16h buildA_perm(const float* M, int ldk,
                                                   int rowBase, int kHalf,
                                                   int lane, int nrows) {
  int r  = rowBase + (lane & 15);
  int kb = kHalf * 32 + ((lane & 16) ? 8 : 0);
  v16h a;
#pragma unroll
  for (int e = 0; e < 16; ++e) {
    int k   = kb + (e < 8 ? e : e + 8);
    int kp  = k;
    float v = (r < nrows) ? M[r * ldk + kp] : 0.0f;
    a[e] = (_Float16)v;
  }
  return a;
}

// f32 A 16x4 layout: lanes 0-15 hold K=0 (VGPR0) / K=1 (VGPR1);
// lanes 16-31 hold K=2 / K=3. We use K = {w0, w1, bias, 0}.
static __device__ __forceinline__ v2f buildA1f32(const float* W1 /*[64][4]*/,
                                                 int rowBase, int lane) {
  int r = rowBase + (lane & 15);
  bool lo = (lane & 16) == 0;
  v2f a;
  a.x = lo ? W1[r * 4 + 0] : W1[r * 4 + 2];
  a.y = lo ? W1[r * 4 + 1] : 0.0f;
  return a;
}

static __device__ __forceinline__ v8f buildC(const float* b, int mBase,
                                             int lane) {
  v8f c;
  int off = mBase + ((lane & 16) ? 8 : 0);
#pragma unroll
  for (int v = 0; v < 8; ++v) c[v] = b[off + v];
  return c;
}

// One MLP evaluation for 16 batch rows; outputs valid in lanes 0-15.
// HASC: layer-2 bias supplied as f32 C tiles (F net).
// !HASC: layer-2 bias rides A2 column phys-63 against a forced constant-1
//        hidden row (G net, whose rows 52..63 are padding).
// ROW0: which D rows of the shared layer-3 tile hold this net's outputs.
template <bool HASC, int ROW0>
static __device__ __forceinline__ void evalNet(const NetCore& N,
                                               const v8f CB2[4],
                                               const v16h A3[2], v2f b1,
                                               float b30, float b31,
                                               float& o0, float& o1) {
  v8f H[4];
#pragma unroll
  for (int t = 0; t < 4; ++t) {
    v8f z = {};
    v8f h = wmma_f32k4(N.A1[t], b1, z);
#pragma unroll
    for (int v = 0; v < 8; ++v) h[v] = act_tanh(h[v]);
    H[t] = h;
  }
  if (!HASC) H[3][7] = 1.0f;  // phys rows 55 & 63 := 1 (bias row; col 55 is 0)

  v16h B2a = packB(H[0], H[1]);
  v16h B2b = packB(H[2], H[3]);

#pragma unroll
  for (int t = 0; t < 4; ++t) {
    v8f acc;
    if constexpr (HASC) acc = CB2[t];
    else                acc = (v8f){};
    acc = wmma_f16(N.A2[t][0], B2a, acc);
    acc = wmma_f16(N.A2[t][1], B2b, acc);
#pragma unroll
    for (int v = 0; v < 8; ++v) acc[v] = act_softplus(acc[v]);
    H[t] = acc;
  }
  v16h B3a = packB(H[0], H[1]);
  v16h B3b = packB(H[2], H[3]);

  v8f z = {};
  v8f O = wmma_f16(A3[0], B3a, z);
  O = wmma_f16(A3[1], B3b, O);
  o0 = O[ROW0 + 0] + b30;   // row ROW0:   vgpr ROW0,   lanes 0-15
  o1 = O[ROW0 + 1] + b31;   // row ROW0+1: vgpr ROW0+1, lanes 0-15
}

// ---- kernel --------------------------------------------------------------

__global__ void __launch_bounds__(256, 1)
csnode_rollout(const float* __restrict__ y0in, const float* __restrict__ tarr,
               const float* __restrict__ f_w1, const float* __restrict__ f_b1,
               const float* __restrict__ f_w2, const float* __restrict__ f_b2,
               const float* __restrict__ f_w3, const float* __restrict__ f_b3,
               const float* __restrict__ g_w1, const float* __restrict__ g_b1,
               const float* __restrict__ g_w2, const float* __restrict__ g_b2,
               const float* __restrict__ g_w3, const float* __restrict__ g_b3,
               float* __restrict__ out) {
  __shared__ float sW1[2][64][4];   // (w0, w1, b1, 0) per hidden row
  __shared__ float sW2[2][64][64];  // zero-padded; [1][m][63] = g_b2[m]
  __shared__ float sW3[4][64];      // rows 0,1 = f_w3; rows 2,3 = g_w3 (padded)
  __shared__ float sB2f[64];        // f_b2 (f net keeps f32 C tiles)

  const int tid = threadIdx.x;
  for (int i = tid; i < 2 * 64 * 64; i += 256) (&sW2[0][0][0])[i] = 0.0f;
  for (int i = tid; i < 2 * 64 * 4;  i += 256) (&sW1[0][0][0])[i] = 0.0f;
  for (int i = tid; i < 4 * 64;      i += 256) (&sW3[0][0])[i]    = 0.0f;
  __syncthreads();

  for (int m = tid; m < 64; m += 256) {
    sW1[0][m][0] = f_w1[2 * m];
    sW1[0][m][1] = f_w1[2 * m + 1];
    sW1[0][m][2] = f_b1[m];
    sB2f[m]      = f_b2[m];
  }
  for (int i = tid; i < 64 * 64; i += 256) sW2[0][i / 64][i % 64] = f_w2[i];
  for (int i = tid; i < 2 * 64;  i += 256) sW3[i / 64][i % 64]    = f_w3[i];

  for (int m = tid; m < 52; m += 256) {
    sW1[1][m][0] = g_w1[2 * m];
    sW1[1][m][1] = g_w1[2 * m + 1];
    sW1[1][m][2] = g_b1[m];
    sW2[1][m][63] = g_b2[m];        // bias column vs forced constant-1 row 63
  }
  for (int i = tid; i < 52 * 52; i += 256) sW2[1][i / 52][i % 52]   = g_w2[i];
  for (int i = tid; i < 2 * 52;  i += 256) sW3[2 + i / 52][i % 52]  = g_w3[i];
  __syncthreads();

  const int  lane = tid & 31;
  const bool lo   = (lane & 16) == 0;
  const long base = ((long)blockIdx.x * 8 + (tid >> 5)) * 16; // 16 rows / wave

  NetCore F, G;
  v8f  CB2f[4];
  v16h A3[2];
#pragma unroll
  for (int t = 0; t < 4; ++t) {
    F.A1[t] = buildA1f32(&sW1[0][0][0], 16 * t, lane);
    G.A1[t] = buildA1f32(&sW1[1][0][0], 16 * t, lane);
    CB2f[t] = buildC(&sB2f[0], 16 * t, lane);
#pragma unroll
    for (int h = 0; h < 2; ++h) {
      F.A2[t][h] = buildA_perm(&sW2[0][0][0], 64, 16 * t, h, lane, 64);
      G.A2[t][h] = buildA_perm(&sW2[1][0][0], 64, 16 * t, h, lane, 64);
    }
  }
#pragma unroll
  for (int h = 0; h < 2; ++h)
    A3[h] = buildA_perm(&sW3[0][0], 64, 0, h, lane, 4); // rows: f0,f1,g0,g1

  const float fb30 = f_b3[0], fb31 = f_b3[1];
  const float gb30 = g_b3[0], gb31 = g_b3[1];

  const int n = lane & 15;                 // batch row within the wave's tile
  float yv0 = y0in[(base + n) * 2 + 0];
  float yv1 = y0in[(base + n) * 2 + 1];
  const float dt = tarr[1] - tarr[0];

  if (lo) {
    v2f y; y.x = yv0; y.y = yv1;
    __builtin_nontemporal_store(y, (v2f*)(out + (base + n) * 2)); // t = 0
  }

  for (int s = 1; s < TSTEPS; ++s) {
    // Layer-1 B (f32 4x16): lanes 0-15 carry K=0/K=1 = (y0, y1);
    // lanes 16-31 carry K=2/K=3 = (1, 0) -> bias row.
    v2f b1;
    b1.x = lo ? yv0 : 1.0f;
    b1.y = lo ? yv1 : 0.0f;

    float u0, u1, f0, f1;
    evalNet<false, 2>(G, CB2f, A3, b1, gb30, gb31, u0, u1);
    evalNet<true,  0>(F, CB2f, A3, b1, fb30, fb31, f0, f1);
    yv0 += dt * (f0 + u0);
    yv1 += dt * (f1 + u1);
    if (lo) {
      v2f y; y.x = yv0; y.y = yv1;
      __builtin_nontemporal_store(
          y, (v2f*)(out + ((size_t)s * NBATCH + base + n) * 2));
    }
  }
}

// ---- launch --------------------------------------------------------------

extern "C" void kernel_launch(void* const* d_in, const int* in_sizes, int n_in,
                              void* d_out, int out_size, void* d_ws,
                              size_t ws_size, hipStream_t stream) {
  (void)in_sizes; (void)n_in; (void)out_size; (void)d_ws; (void)ws_size;
  const float* y0   = (const float*)d_in[0];
  const float* t    = (const float*)d_in[1];
  const float* f_w1 = (const float*)d_in[2];
  const float* f_b1 = (const float*)d_in[3];
  const float* f_w2 = (const float*)d_in[4];
  const float* f_b2 = (const float*)d_in[5];
  const float* f_w3 = (const float*)d_in[6];
  const float* f_b3 = (const float*)d_in[7];
  const float* g_w1 = (const float*)d_in[8];
  const float* g_b1 = (const float*)d_in[9];
  const float* g_w2 = (const float*)d_in[10];
  const float* g_b2 = (const float*)d_in[11];
  const float* g_w3 = (const float*)d_in[12];
  const float* g_b3 = (const float*)d_in[13];

  // 65536 rows / 16 per wave / 8 waves per block = 512 blocks
  dim3 grid(NBATCH / (16 * 8));
  dim3 block(256);
  csnode_rollout<<<grid, block, 0, stream>>>(
      y0, t, f_w1, f_b1, f_w2, f_b2, f_w3, f_b3,
      g_w1, g_b1, g_w2, g_b2, g_w3, g_b3, (float*)d_out);
}
